// VQCLayer_1494648619142
// MI455X (gfx1250) — matrix operand order repaired and checked by reading, not compiled, not run on hardware
//
#include <hip/hip_runtime.h>
#include <hip/hip_bf16.h>

typedef __attribute__((ext_vector_type(16))) _Float16 v16h;
typedef __attribute__((ext_vector_type(8)))  float    v8f;

#define NQ   8
#define DIM  256                  // 2^8 basis states
#define KT_N 8                    // K tiles: 256 / 32
#define MT_N 16                   // M tiles: 256 / 16
#define APACK_HALVES 65536        // halves per component: 16*8*32*16
#define APACK_TOTAL  131072       // halves total (re|im) = 256 KB

// ---------------------------------------------------------------------------
// Kernel 1: build the variational unitary U (256x256 complex) from weights.
// Block j simulates basis state e_j through 2x (CNOT chain + per-qubit RX),
// thread m owns amplitude m. Result stored as f16 re/im in a WMMA
// A-fragment layout with contiguous k (k = 16*(lane/16) + element).
// ---------------------------------------------------------------------------
__global__ __launch_bounds__(256) void vqc_build_unitary(
    const float* __restrict__ w, _Float16* __restrict__ Apack)
{
  __shared__ float lre[DIM];
  __shared__ float lim[DIM];
  const int m = threadIdx.x;   // amplitude index owned by this thread
  const int j = blockIdx.x;    // column of U (initial basis state)

  float ar = (m == j) ? 1.0f : 0.0f;
  float ai = 0.0f;

  for (int layer = 0; layer < 2; ++layer) {
    // CNOT chain (0->1, ..., 6->7) == basis permutation g.
    // new[m] = old[g^{-1}(m)];  g^{-1} applies CNOT(6,7) first, CNOT(0,1) last.
    lre[m] = ar; lim[m] = ai;
    __syncthreads();
    int src = m;
    #pragma unroll
    for (int i = 6; i >= 0; --i)
      src ^= ((src >> i) & 1) << (i + 1);
    ar = lre[src]; ai = lim[src];
    __syncthreads();

    // RX(w[layer][q]) on each qubit: new = c*self - i*s*partner
    for (int q = 0; q < NQ; ++q) {
      const float th = 0.5f * w[layer * NQ + q];
      const float c = cosf(th), s = sinf(th);
      lre[m] = ar; lim[m] = ai;
      __syncthreads();
      const int p = m ^ (1 << q);
      const float pr = lre[p], pi = lim[p];
      const float nr = c * ar + s * pi;
      const float ni = c * ai - s * pr;
      ar = nr; ai = ni;
      __syncthreads();
    }
  }

  // Scatter U[m][j] into the f16 A-fragment layout:
  // lanes 0-15 hold K 0..15; lanes 16-31 hold K 16..31.
  const int mt = m >> 4, mr = m & 15;
  const int kt = j >> 5, kr = j & 31;
  const int lane = mr + 16 * (kr >> 4);
  const int h    = kr & 15;
  const size_t idx = ((size_t)(mt * KT_N + kt) * 32 + lane) * 16 + h;
  Apack[idx]                = (_Float16)ar;
  Apack[APACK_HALVES + idx] = (_Float16)ai;
}

// ---------------------------------------------------------------------------
// Kernel 2: 8 waves per workgroup, 16 samples per wave (128 per block).
// Step 1: async-stage the full packed U (256 KB) into LDS once per block
//         (global_load_async_to_lds_b128 + s_wait_asynccnt), shared by all
//         8 waves -> 8x less L2 traffic than per-wave global reads.
// Step 2: per wave, Y = U * Psi0 (two real f16->f32 WMMA GEMMs, A from LDS,
//         B built on the fly from cos/sin products).
// Step 3: sign-weighted |Y|^2 reduction via Walsh-Hadamard butterfly.
// ---------------------------------------------------------------------------
__global__ __launch_bounds__(256) void vqc_wmma(
    const float* __restrict__ x, const _Float16* __restrict__ Apack,
    float* __restrict__ out)
{
  __shared__ __align__(32) _Float16 Ash[APACK_TOTAL];   // 256 KB (<= 320 KB/WGP)

  const int tid   = threadIdx.x;
  const int lane  = tid & 31;
  const int wave  = tid >> 5;
  const int ntile = blockIdx.x * 8 + wave;
  const int col   = ntile * 16 + (lane & 15);   // batch sample for this lane
  const int hi    = lane >> 4;                  // 0: K 0-15 half, 1: K 16-31 half

  // ---- Stage U -> LDS with the CDNA5 async copy path (ASYNCcnt) ----------
  {
    const char* gsrc = (const char*)Apack;
    for (int i = tid; i < APACK_TOTAL / 8; i += 256) {  // 16384 x 16B chunks
      const unsigned ldsAddr = (unsigned)(uintptr_t)&Ash[(size_t)i * 8];
      const void*    src     = gsrc + (size_t)i * 16;
      asm volatile("global_load_async_to_lds_b128 %0, %1, off"
                   :: "v"(ldsAddr), "v"(src) : "memory");
    }
    asm volatile("s_wait_asynccnt 0" ::: "memory");
  }
  __syncthreads();

  // ---- Encoding angles -> cos/sin; psi0 = plo[low nibble]*phi[high nibble]
  float c[NQ], s[NQ];
  #pragma unroll
  for (int i = 0; i < NQ; ++i) {
    const float t = 0.5f * x[(size_t)col * NQ + i];
    c[i] = cosf(t);
    s[i] = sinf(t);
  }
  float plo[16], phi[16];
  #pragma unroll
  for (int h = 0; h < 16; ++h) {
    plo[h] = ((h & 1) ? s[0] : c[0]) * ((h & 2) ? s[1] : c[1]) *
             ((h & 4) ? s[2] : c[2]) * ((h & 8) ? s[3] : c[3]);
    phi[h] = ((h & 1) ? s[4] : c[4]) * ((h & 2) ? s[5] : c[5]) *
             ((h & 4) ? s[6] : c[6]) * ((h & 8) ? s[7] : c[7]);
  }

  v8f zero = {};
  v8f accR[MT_N], accI[MT_N];
  #pragma unroll
  for (int mt = 0; mt < MT_N; ++mt) { accR[mt] = zero; accI[mt] = zero; }

  // ---- Main WMMA loop: A fragments from LDS, B generated per k-step ------
  #pragma unroll
  for (int kt = 0; kt < KT_N; ++kt) {
    const float pg = phi[(kt << 1) | hi];
    v16h b;
    #pragma unroll
    for (int h = 0; h < 16; ++h) b[h] = (_Float16)(plo[h] * pg);

    const size_t aOff = ((size_t)kt * 32 + lane) * 16;
    #pragma unroll
    for (int mt = 0; mt < MT_N; ++mt) {
      const size_t o = aOff + (size_t)mt * (KT_N * 32 * 16);
      v16h are = *(const v16h*)&Ash[o];
      v16h aim = *(const v16h*)&Ash[o + APACK_HALVES];
      accR[mt] = __builtin_amdgcn_wmma_f32_16x16x32_f16(
          false, are, false, b, (short)0, accR[mt], false, false);
      accI[mt] = __builtin_amdgcn_wmma_f32_16x16x32_f16(
          false, aim, false, b, (short)0, accI[mt], false, false);
    }
  }

  // ---- Epilogue: sign-weighted probability reduction (WHT butterfly) -----
  // Lane L holds (M = mt*16 + 8*hi + v, N = L&15) in acc[mt][v].
  // m bits: 0-2 = v, 3 = hi, 4-7 = mt.
  float Sv[MT_N];
  float D0 = 0.0f, D1 = 0.0f, D2 = 0.0f;
  #pragma unroll
  for (int mt = 0; mt < MT_N; ++mt) {
    float p[8];
    #pragma unroll
    for (int v = 0; v < 8; ++v) {
      const float yr = accR[mt][v];
      const float yi = accI[mt][v];
      p[v] = yr * yr + yi * yi;
    }
    const float a01 = p[0] + p[1], a23 = p[2] + p[3];
    const float a45 = p[4] + p[5], a67 = p[6] + p[7];
    D0 += ((p[0] - p[1]) + (p[2] - p[3])) + ((p[4] - p[5]) + (p[6] - p[7]));
    D1 += (a01 - a23) + (a45 - a67);
    const float s03 = a01 + a23, s47 = a45 + a67;
    D2 += s03 - s47;
    Sv[mt] = s03 + s47;
  }
  // Butterfly over the mt axis (m bits 4-7).
  float e0[8], P4 = 0.0f;
  #pragma unroll
  for (int jj = 0; jj < 8; ++jj) {
    e0[jj] = Sv[2 * jj] + Sv[2 * jj + 1];
    P4    += Sv[2 * jj] - Sv[2 * jj + 1];
  }
  float e1[4], P5 = 0.0f;
  #pragma unroll
  for (int jj = 0; jj < 4; ++jj) {
    e1[jj] = e0[2 * jj] + e0[2 * jj + 1];
    P5    += e0[2 * jj] - e0[2 * jj + 1];
  }
  const float P6   = (e1[0] - e1[1]) + (e1[2] - e1[3]);
  const float f0   = e1[0] + e1[1], f1 = e1[2] + e1[3];
  const float P7   = f0 - f1;
  const float Stot = f0 + f1;

  float part[NQ] = { D0, D1, D2, hi ? -Stot : Stot, P4, P5, P6, P7 };
  // Combine the two half-lane groups (complementary M sets, same column).
  #pragma unroll
  for (int i = 0; i < NQ; ++i)
    part[i] += __shfl_xor(part[i], 16, 32);

  if (hi == 0) {
    #pragma unroll
    for (int i = 0; i < NQ; ++i)
      out[(size_t)col * NQ + i] = part[i];
  }
}

extern "C" void kernel_launch(void* const* d_in, const int* in_sizes, int n_in,
                              void* d_out, int out_size, void* d_ws, size_t ws_size,
                              hipStream_t stream) {
  const float* inputs  = (const float*)d_in[0];   // (B, 8) float32
  const float* weights = (const float*)d_in[1];   // (2, 8) float32
  float*       out     = (float*)d_out;           // (B, 8) float32
  _Float16*    Apack   = (_Float16*)d_ws;         // 256 KB packed U (re|im)

  const int B = in_sizes[0] / NQ;                 // 65536

  // Build U from weights, then the batched WMMA contraction.
  vqc_build_unitary<<<DIM, DIM, 0, stream>>>(weights, Apack);
  // 8 waves x 16 samples = 128 samples per block.
  vqc_wmma<<<B / 128, 256, 0, stream>>>(inputs, Apack, out);
}
